// MultiHeadSelfAttention_3324304687799
// MI455X (gfx1250) — compile-verified, matrix operand order checked by reading, not run on hardware
//
#include <hip/hip_runtime.h>
#include <hip/hip_bf16.h>

typedef __attribute__((ext_vector_type(16))) _Float16 v16h;
typedef __attribute__((ext_vector_type(8)))  _Float16 v8h;
typedef __attribute__((ext_vector_type(8)))  float    v8f;
typedef __attribute__((ext_vector_type(4)))  int      v4i;

#define B_  2
#define T_  2048
#define D_  1024
#define H_  16
#define DH_ 64
#define LN_EPS 1e-3f

#if __has_builtin(__builtin_amdgcn_global_load_async_to_lds_b128)
#define HAS_ASYNC_LDS 1
#else
#define HAS_ASYNC_LDS 0
#endif

#if HAS_ASYNC_LDS
// GLOBAL_LOAD_ASYNC_TO_LDS_B128: same imm offset applies to global and LDS addr.
// Toolchain declares both pointer params as generic non-const v4i*.
#define ASYNC_B128(g, l, off)                                              \
    __builtin_amdgcn_global_load_async_to_lds_b128(                        \
        (v4i*)(g), (v4i*)(l), (off), 0)
#endif

#if __has_builtin(__builtin_amdgcn_s_wait_asynccnt)
#define WAIT_ASYNC(n) __builtin_amdgcn_s_wait_asynccnt(n)
#elif HAS_ASYNC_LDS
#define WAIT_ASYNC(n) asm volatile("s_wait_asynccnt %0" ::"i"(n) : "memory")
#else
#define WAIT_ASYNC(n) ((void)0)
#endif

// ---------------------------------------------------------------------------
// Fragment helpers (CDNA5 16-bit A-matrix layout, wave32):
//   lane 0-15  : row m = lane,     VGPRs hold K = {0..7, 16..23}
//   lane 16-31 : row m = lane-16,  VGPRs hold K = {8..15, 24..31}
// B fragments use the mirrored layout with n = lane%16, reading from a
// transposed (N-major) tile so per-lane reads are contiguous.
// ---------------------------------------------------------------------------
__device__ inline v16h make_frag(v8h lo, v8h hi) {
    union { v16h v; v8h h[2]; } u;
    u.h[0] = lo; u.h[1] = hi;
    return u.v;
}

__device__ inline v16h load_frag(const _Float16* p, int khi) {
    v8h lo = *(const v8h*)(p + khi);
    v8h hi = *(const v8h*)(p + 16 + khi);
    return make_frag(lo, hi);
}

// ---------------------------------------------------------------------------
// Kernel 0: weight transpose + f32 -> f16 convert.  w[Din][Nout] -> wt[Nout][Din]
// ---------------------------------------------------------------------------
__global__ __launch_bounds__(256)
void wt_kernel(const float* __restrict__ w, _Float16* __restrict__ wt,
               int Din, int Nout) {
    int idx = blockIdx.x * 256 + threadIdx.x;
    int n = idx / Din;
    int k = idx - n * Din;
    wt[(size_t)n * Din + k] = (_Float16)w[(size_t)k * Nout + n];
}

// ---------------------------------------------------------------------------
// Kernel 1: LayerNorm row (D=1024) -> f16.  One block per row, 256 threads.
// ---------------------------------------------------------------------------
__global__ __launch_bounds__(256)
void ln_kernel(const float* __restrict__ x, const float* __restrict__ gamma,
               const float* __restrict__ beta, _Float16* __restrict__ h) {
    const int row = blockIdx.x;
    const float* xr = x + (size_t)row * D_;
    float s = 0.f, ss = 0.f;
    float vals[4];
#pragma unroll
    for (int i = 0; i < 4; ++i) {
        float v = xr[threadIdx.x + i * 256];
        vals[i] = v;
        s += v; ss += v * v;
    }
#pragma unroll
    for (int off = 16; off >= 1; off >>= 1) {
        s  += __shfl_xor(s,  off, 32);
        ss += __shfl_xor(ss, off, 32);
    }
    __shared__ float reds[8], redss[8];
    int wave = threadIdx.x >> 5, lane = threadIdx.x & 31;
    if (lane == 0) { reds[wave] = s; redss[wave] = ss; }
    __syncthreads();
    float ts = 0.f, tss = 0.f;
#pragma unroll
    for (int i = 0; i < 8; ++i) { ts += reds[i]; tss += redss[i]; }
    float mean = ts * (1.0f / D_);
    float var  = tss * (1.0f / D_) - mean * mean;
    float rstd = rsqrtf(var + LN_EPS);
    _Float16* hr = h + (size_t)row * D_;
#pragma unroll
    for (int i = 0; i < 4; ++i) {
        int c = threadIdx.x + i * 256;
        hr[c] = (_Float16)((vals[i] - mean) * rstd * gamma[c] + beta[c]);
    }
}

// ---------------------------------------------------------------------------
// Kernel 2/4: tiled WMMA GEMM.  C[M,N] = A[M,K] * Bt[N,K]^T, f32 accumulate.
//   Block: 256 threads = 8 waves, 128x128 tile, BK = 32, double-buffered LDS
//   staged with GLOBAL_LOAD_ASYNC_TO_LDS_B128 (ASYNCcnt pipelined).
//   K is compile-time (= D_) so the s_wait_asynccnt immediates stay literal.
//   mode 0: epilogue scatters f16 into qkv[3][B][H][T][DH], Q scaled 1/sqrt(DH)
//   mode 1: epilogue adds residual x and writes f32 out
// ---------------------------------------------------------------------------
__global__ __launch_bounds__(256)
void gemm_kernel(const _Float16* __restrict__ A, const _Float16* __restrict__ Bt,
                 int mode,
                 _Float16* __restrict__ qkvout,
                 float* __restrict__ outres, const float* __restrict__ xres) {
    constexpr int K = D_;
    __shared__ _Float16 As[2][128][32];
    __shared__ _Float16 Bs[2][128][32];

    const int bm = blockIdx.y * 128;
    const int bn = blockIdx.x * 128;
    const int wave = threadIdx.x >> 5;
    const int lane = threadIdx.x & 31;
    const int wm = wave & 1;        // 2 waves over M (64 rows each)
    const int wn = wave >> 1;       // 4 waves over N (32 cols each)
    const int ml  = lane & 15;
    const int hf  = lane >> 4;
    const int khi = hf * 8;

    const int lrow = threadIdx.x >> 1;
    const int lhl  = threadIdx.x & 1;

    const _Float16* gA0 = A  + (size_t)(bm + lrow) * K + lhl * 16;
    const _Float16* gB0 = Bt + (size_t)(bn + lrow) * K + lhl * 16;

    v8f acc[4][2] = {};

    // ---- stage k-slice kk into LDS buffer `buf` -----------------------------
    auto stage = [&](int kk, int buf) {
        const _Float16* ga = gA0 + kk;
        const _Float16* gb = gB0 + kk;
        _Float16* la = &As[buf][lrow][lhl * 16];
        _Float16* lb = &Bs[buf][lrow][lhl * 16];
#if HAS_ASYNC_LDS
        ASYNC_B128(ga, la, 0);
        ASYNC_B128(ga, la, 16);
        ASYNC_B128(gb, lb, 0);
        ASYNC_B128(gb, lb, 16);
#else
        *(v8h*)la       = *(const v8h*)ga;
        *(v8h*)(la + 8) = *(const v8h*)(ga + 8);
        *(v8h*)lb       = *(const v8h*)gb;
        *(v8h*)(lb + 8) = *(const v8h*)(gb + 8);
#endif
    };

    auto compute = [&](int buf) {
        v16h af[4], bf[2];
#pragma unroll
        for (int mt = 0; mt < 4; ++mt)
            af[mt] = load_frag(&As[buf][wm * 64 + mt * 16 + ml][0], khi);
#pragma unroll
        for (int nt = 0; nt < 2; ++nt)
            bf[nt] = load_frag(&Bs[buf][wn * 32 + nt * 16 + ml][0], khi);
#pragma unroll
        for (int mt = 0; mt < 4; ++mt)
#pragma unroll
            for (int nt = 0; nt < 2; ++nt)
                acc[mt][nt] = __builtin_amdgcn_wmma_f32_16x16x32_f16(
                    false, af[mt], false, bf[nt], (short)0, acc[mt][nt],
                    false, false);
    };

    int buf = 0;
    stage(0, 0);
    for (int kk = 0; kk < K - 32; kk += 32) {
        stage(kk + 32, buf ^ 1);   // prefetch next slice into other buffer
        WAIT_ASYNC(4);             // oldest 4 (current buf) have landed
        __syncthreads();           // make all waves' DMA visible
        compute(buf);
        __syncthreads();           // done reading buf; it may be restaged
        buf ^= 1;
    }
    WAIT_ASYNC(0);                 // drain final slice
    __syncthreads();
    compute(buf);

    // Epilogue: C/D layout -> (m,n): vgpr r, lane l :
    //   m = r + 8*(l/16), n = l%16  (within each 16x16 tile)
    const int nl = ml;
#pragma unroll
    for (int mt = 0; mt < 4; ++mt) {
#pragma unroll
        for (int nt = 0; nt < 2; ++nt) {
#pragma unroll
            for (int r = 0; r < 8; ++r) {
                int Rr = bm + wm * 64 + mt * 16 + r + 8 * hf;
                int Cc = bn + wn * 32 + nt * 16 + nl;
                float v = acc[mt][nt][r];
                if (mode == 0) {
                    int i   = Cc >> 10;
                    int rem = Cc & 1023;
                    int hh  = rem >> 6;
                    int dh  = rem & 63;
                    int bb  = Rr >> 11;
                    int tt  = Rr & 2047;
                    if (i == 0) v *= 0.125f;   // 1/sqrt(DH)
                    qkvout[((((size_t)i * B_ + bb) * H_ + hh) * T_ + tt) * DH_ + dh]
                        = (_Float16)v;
                } else {
                    size_t off = (size_t)Rr * D_ + Cc;
                    outres[off] = v + xres[off];
                }
            }
        }
    }
}

// ---------------------------------------------------------------------------
// Kernel 3: flash attention.  grid = (T/128, H, B), 256 threads = 8 waves.
// Wave owns 16 query rows; loops over key chunks of 128 with online softmax.
// K chunk staged via async DMA to LDS; V transposed through VGPRs.
// ---------------------------------------------------------------------------
__global__ __launch_bounds__(256)
void attn_kernel(const _Float16* __restrict__ qkv,
                 _Float16* __restrict__ attnout) {
    __shared__ _Float16 Ks[128][64];    // key chunk, row-major  (16 KB)
    __shared__ _Float16 Vt[64][128];    // value chunk, transposed (16 KB)
    __shared__ _Float16 Ps[128][128];   // probs, per-wave 16 rows (32 KB)

    const int qc = blockIdx.x;          // query chunk
    const int hh = blockIdx.y;
    const int bb = blockIdx.z;

    const _Float16* Qp = qkv + (((size_t)0 * B_ + bb) * H_ + hh) * (size_t)T_ * DH_;
    const _Float16* Kp = qkv + (((size_t)1 * B_ + bb) * H_ + hh) * (size_t)T_ * DH_;
    const _Float16* Vp = qkv + (((size_t)2 * B_ + bb) * H_ + hh) * (size_t)T_ * DH_;

    const int wave = threadIdx.x >> 5;
    const int lane = threadIdx.x & 31;
    const int ml   = lane & 15;
    const int hf   = lane >> 4;
    const int khi  = hf * 8;

    // Q fragments (held in registers for the whole kernel); row = query
    const int q = qc * 128 + wave * 16 + ml;
    v16h qf[2];
#pragma unroll
    for (int s = 0; s < 2; ++s)
        qf[s] = load_frag(Qp + (size_t)q * DH_ + s * 32, khi);

    float mrow[8], lsum[8];
    v8f accO[4] = {};
#pragma unroll
    for (int r = 0; r < 8; ++r) { mrow[r] = -1e30f; lsum[r] = 0.f; }

    const int lkey = threadIdx.x >> 1;
    const int lhl  = threadIdx.x & 1;

    for (int c = 0; c < T_; c += 128) {
        // K chunk: async DMA straight into LDS (no VGPR round-trip)
        {
            const _Float16* gk = Kp + (size_t)(c + lkey) * DH_ + lhl * 32;
            _Float16* lk = &Ks[lkey][lhl * 32];
#if HAS_ASYNC_LDS
            ASYNC_B128(gk, lk, 0);
            ASYNC_B128(gk, lk, 16);
            ASYNC_B128(gk, lk, 32);
            ASYNC_B128(gk, lk, 48);
#else
#pragma unroll
            for (int jj = 0; jj < 4; ++jj)
                *(v8h*)(lk + jj * 8) = *(const v8h*)(gk + jj * 8);
#endif
        }
        // V chunk: transpose through VGPRs (overlaps with the K DMA)
#pragma unroll
        for (int jj = 0; jj < 4; ++jj) {
            v8h tv = *(const v8h*)(Vp + (size_t)(c + lkey) * DH_ + lhl * 32 + jj * 8);
#pragma unroll
            for (int j2 = 0; j2 < 8; ++j2)
                Vt[lhl * 32 + jj * 8 + j2][lkey] = tv[j2];
        }
        WAIT_ASYNC(0);
        __syncthreads();

        // S = Q * K^T   (16 WMMAs per wave: 2 k-steps x 8 n-tiles)
        v8f sacc[8] = {};
#pragma unroll
        for (int s = 0; s < 2; ++s) {
#pragma unroll
            for (int nt = 0; nt < 8; ++nt) {
                v16h bk = load_frag(&Ks[nt * 16 + ml][s * 32], khi);
                sacc[nt] = __builtin_amdgcn_wmma_f32_16x16x32_f16(
                    false, qf[s], false, bk, (short)0, sacc[nt], false, false);
            }
        }

        // online softmax (per row r; xor masks 1/2/4/8 stay in each 16-lane half)
#pragma unroll
        for (int r = 0; r < 8; ++r) {
            float mx = sacc[0][r];
#pragma unroll
            for (int nt = 1; nt < 8; ++nt) mx = fmaxf(mx, sacc[nt][r]);
#pragma unroll
            for (int off = 1; off < 16; off <<= 1)
                mx = fmaxf(mx, __shfl_xor(mx, off, 32));
            float mnew  = fmaxf(mrow[r], mx);
            float alpha = __expf(mrow[r] - mnew);
            mrow[r] = mnew;
            float rs = 0.f;
#pragma unroll
            for (int nt = 0; nt < 8; ++nt) {
                float p = __expf(sacc[nt][r] - mnew);
                sacc[nt][r] = p;
                rs += p;
            }
#pragma unroll
            for (int off = 1; off < 16; off <<= 1)
                rs += __shfl_xor(rs, off, 32);
            lsum[r] = lsum[r] * alpha + rs;
#pragma unroll
            for (int ot = 0; ot < 4; ++ot) accO[ot][r] *= alpha;
        }

        // spill P (f16) to LDS in row-major so it can be re-read as A fragments
#pragma unroll
        for (int nt = 0; nt < 8; ++nt)
#pragma unroll
            for (int r = 0; r < 8; ++r)
                Ps[wave * 16 + r + 8 * hf][nt * 16 + ml] = (_Float16)sacc[nt][r];

        // O += P * V    (16 WMMAs per wave: 4 k-steps x 4 n-tiles)
#pragma unroll
        for (int s2 = 0; s2 < 4; ++s2) {
            v16h ap = load_frag(&Ps[wave * 16 + ml][s2 * 32], khi);
#pragma unroll
            for (int ot = 0; ot < 4; ++ot) {
                v16h bv = load_frag(&Vt[ot * 16 + ml][s2 * 32], khi);
                accO[ot] = __builtin_amdgcn_wmma_f32_16x16x32_f16(
                    false, ap, false, bv, (short)0, accO[ot], false, false);
            }
        }
        __syncthreads();
    }

    // normalize + write [B,T,H*DH] as f16 for the output projection
#pragma unroll
    for (int r = 0; r < 8; ++r) {
        float inv = 1.0f / lsum[r];
        int tq = qc * 128 + wave * 16 + r + 8 * hf;
#pragma unroll
        for (int ot = 0; ot < 4; ++ot) {
            int dh = ot * 16 + ml;
            attnout[((size_t)bb * T_ + tq) * D_ + hh * DH_ + dh] =
                (_Float16)(accO[ot][r] * inv);
        }
    }
}

// ---------------------------------------------------------------------------
// Host-side launch
// ---------------------------------------------------------------------------
extern "C" void kernel_launch(void* const* d_in, const int* in_sizes, int n_in,
                              void* d_out, int out_size, void* d_ws, size_t ws_size,
                              hipStream_t stream) {
    const float* x     = (const float*)d_in[0];
    const float* gamma = (const float*)d_in[1];
    const float* beta  = (const float*)d_in[2];
    const float* w_qkv = (const float*)d_in[3];
    const float* w_out = (const float*)d_in[4];
    float* out = (float*)d_out;

    char* ws = (char*)d_ws;
    _Float16* h16     = (_Float16*)(ws);                       //  8 MiB
    _Float16* wqkvT   = (_Float16*)(ws + (8u  << 20));         //  6 MiB
    _Float16* woutT   = (_Float16*)(ws + (14u << 20));         //  2 MiB
    _Float16* qkvbuf  = (_Float16*)(ws + (16u << 20));         // 24 MiB
    _Float16* attnout = (_Float16*)(ws + (40u << 20));         //  8 MiB

    // weights: f32 -> f16, transposed to [N][K]
    wt_kernel<<<(3 * D_ * D_) / 256, 256, 0, stream>>>(w_qkv, wqkvT, D_, 3 * D_);
    wt_kernel<<<(D_ * D_) / 256, 256, 0, stream>>>(w_out, woutT, D_, D_);

    // LayerNorm -> f16 activations
    ln_kernel<<<B_ * T_, 256, 0, stream>>>(x, gamma, beta, h16);

    // QKV projection: [4096,1024] x [1024,3072] -> scattered q/k/v f16
    gemm_kernel<<<dim3((3 * D_) / 128, (B_ * T_) / 128), 256, 0, stream>>>(
        h16, wqkvT, 0, qkvbuf, nullptr, nullptr);

    // flash attention per (b,h,query-chunk)
    attn_kernel<<<dim3(T_ / 128, H_, B_), 256, 0, stream>>>(qkvbuf, attnout);

    // output projection + residual -> f32
    gemm_kernel<<<dim3(D_ / 128, (B_ * T_) / 128), 256, 0, stream>>>(
        attnout, woutT, 1, nullptr, out, x);
}
